// CTCLayer_7645041787031
// MI455X (gfx1250) — compile-verified, matrix operand order checked by reading, not run on hardware
//
#include <hip/hip_runtime.h>
#include <hip/hip_bf16.h>

typedef __attribute__((ext_vector_type(16))) _Float16 v16h;
typedef __attribute__((ext_vector_type(8)))  float    v8f;

#define B_     128
#define T_     512
#define C_     1024
#define LMAX_  64
#define S_     129            // 2*Lmax + 1
#define BLANK_ (C_ - 1)
#define NEGINF (-1e30f)
#define EPSK   (1e-7f)

// XOR-swizzled LDS float index for element (row r, col c) of the staged 16x1024 tile.
// Atoms are 16B (float4); low 4 atom bits are XORed with the row to kill the
// 4KB-row-stride bank aliasing (1024 f32 == 0 mod 64 banks).
__device__ __forceinline__ int lds_idx(int r, int c) {
    int a = c >> 2;
    int as = (a & 0xF0) | ((a ^ r) & 15);
    return r * 1024 + as * 4 + (c & 3);
}
__device__ __forceinline__ int lds_atom(int r, int a) {       // float index of atom start
    return r * 1024 + ((a & 0xF0) | ((a ^ r) & 15)) * 4;
}

// ---------------------------------------------------------------------------
// Kernel 1: per 16 rows (one workgroup, 256 thr / 8 waves):
//   - coalesced stage of 16x1024 f32 into LDS (64 KB, swizzled)
//   - row sums via v_wmma_f32_16x16x32_f16 against an all-ones B (f32 accum)
//   - gather S=129 extended-label entries, write lp_ext[b,t,s]
// ---------------------------------------------------------------------------
__global__ __launch_bounds__(256)
void ctc_rowprep(const float* __restrict__ y_pred,
                 const int*   __restrict__ y_true,
                 float*       __restrict__ lp_ext)
{
    extern __shared__ float lds_rows[];      // 16*1024 floats (64 KB), swizzled
    __shared__ float s_part[128];            // 8 waves x 16 rows partial sums
    __shared__ float s_logrs[16];
    __shared__ int   s_ext[S_];

    const int tid  = threadIdx.x;
    const int lane = tid & 31;
    const int wv   = tid >> 5;               // 0..7

    const long rowBase = (long)blockIdx.x * 16;      // flat row = b*T + t
    const int  b       = (int)(rowBase / T_);        // 16 rows never cross b (T%16==0)
    const float* src   = y_pred + rowBase * C_;

    if (tid < S_)
        s_ext[tid] = (tid & 1) ? y_true[b * LMAX_ + (tid >> 1)] : BLANK_;

    // ---- stage 16 rows, perfectly coalesced 16B vectors ----
    const float4* g4 = (const float4*)src;
    #pragma unroll
    for (int k = 0; k < 16; ++k) {
        int q = k * 256 + tid;               // float4 index 0..4095
        int r = q >> 8;
        int a = q & 255;
        float4 v = g4[q];
        *(float4*)&lds_rows[lds_atom(r, a)] = v;
    }
    __syncthreads();

    // ---- row sums: wave wv reduces columns [wv*128, wv*128+128) via WMMA ----
    v16h onesv;
    #pragma unroll
    for (int j = 0; j < 16; ++j) onesv[j] = (_Float16)1.0f;

    v8f acc = {};
    const int row  = lane & 15;
    const int half = lane >> 4;
    #pragma unroll
    for (int c = 0; c < 4; ++c) {
        const int colbase = wv * 128 + c * 32;
        v16h afrag;
        {   // elements 0..7  : K = 8*half + 0..7
            int a0 = (colbase + 8 * half) >> 2;        // even
            float4 u = *(const float4*)&lds_rows[lds_atom(row, a0)];
            float4 v = *(const float4*)&lds_rows[lds_atom(row, a0 + 1)];
            afrag[0]=(_Float16)u.x; afrag[1]=(_Float16)u.y;
            afrag[2]=(_Float16)u.z; afrag[3]=(_Float16)u.w;
            afrag[4]=(_Float16)v.x; afrag[5]=(_Float16)v.y;
            afrag[6]=(_Float16)v.z; afrag[7]=(_Float16)v.w;
        }
        {   // elements 8..15 : K = 16 + 8*half + 0..7
            int a0 = (colbase + 16 + 8 * half) >> 2;
            float4 u = *(const float4*)&lds_rows[lds_atom(row, a0)];
            float4 v = *(const float4*)&lds_rows[lds_atom(row, a0 + 1)];
            afrag[8] =(_Float16)u.x; afrag[9] =(_Float16)u.y;
            afrag[10]=(_Float16)u.z; afrag[11]=(_Float16)u.w;
            afrag[12]=(_Float16)v.x; afrag[13]=(_Float16)v.y;
            afrag[14]=(_Float16)v.z; afrag[15]=(_Float16)v.w;
        }
        // D[m,n] = sum_k A[m,k]  (every column identical), f32 accumulate
        acc = __builtin_amdgcn_wmma_f32_16x16x32_f16(
                  false, afrag, false, onesv, (short)0, acc, false, false);
    }
    // lane 0 holds column 0 of rows 0..7 in VGPRs 0..7; lane 16 rows 8..15
    if (lane == 0) {
        #pragma unroll
        for (int r2 = 0; r2 < 8; ++r2) s_part[wv * 16 + r2] = acc[r2];
    }
    if (lane == 16) {
        #pragma unroll
        for (int r2 = 0; r2 < 8; ++r2) s_part[wv * 16 + 8 + r2] = acc[r2];
    }
    __syncthreads();

    if (tid < 16) {
        float s = (float)C_ * EPSK;                  // Σ_c (y + eps)
        #pragma unroll
        for (int w = 0; w < 8; ++w) s += s_part[w * 16 + tid];
        s_logrs[tid] = logf(s);
    }
    __syncthreads();

    // ---- gather extended-label log-probs: lp = log(y+eps) - log(rowsum) ----
    for (int i = tid; i < 16 * S_; i += 256) {
        int r = i / S_;
        int s = i - r * S_;
        float y = lds_rows[lds_idx(r, s_ext[s])];
        lp_ext[(rowBase + r) * S_ + s] = logf(y + EPSK) - s_logrs[r];
    }
}

// ---------------------------------------------------------------------------
// Kernel 2: CTC forward DP. One workgroup (160 thr / 5 waves) per batch row.
// Double-buffered alpha in LDS -> one barrier per time step; lp prefetched.
// ---------------------------------------------------------------------------
__global__ __launch_bounds__(160)
void ctc_dp(const float* __restrict__ lp_ext,
            const int*   __restrict__ y_true,
            const int*   __restrict__ y_len,
            float*       __restrict__ out)
{
    __shared__ float abuf[2][S_ + 3];

    const int b = blockIdx.x;
    const int s = threadIdx.x;
    const bool act = (s < S_);

    bool skipf = false;                      // s>=2, odd, ext[s]!=ext[s-2]
    if (act && (s & 1) && s >= 3) {
        int l0 = y_true[b * LMAX_ + (s >> 1)];
        int l1 = y_true[b * LMAX_ + (s >> 1) - 1];
        skipf = (l0 != l1);
    }

    const float* lp = lp_ext + (size_t)b * T_ * S_;

    if (act) abuf[0][s] = (s < 2) ? lp[s] : NEGINF;   // alpha_0
    __syncthreads();

    int prev = 0;
    float lpv = act ? lp[S_ + s] : 0.0f;              // t = 1
    for (int t = 1; t < T_; ++t) {
        float lpn = (act && (t + 1 < T_)) ? lp[(t + 1) * S_ + s] : 0.0f;
        if (act) {
            const float* P = abuf[prev];
            float a1 = P[s];
            float a2 = (s >= 1) ? P[s - 1] : NEGINF;
            float a3 = skipf    ? P[s - 2] : NEGINF;
            float m  = fmaxf(a1, fmaxf(a2, a3));
            float v  = m + logf(expf(a1 - m) + expf(a2 - m) + expf(a3 - m)) + lpv;
            abuf[prev ^ 1][s] = v;
        }
        __syncthreads();
        prev ^= 1;
        lpv = lpn;
    }

    if (s == 0) {
        int L = y_len[b];                              // 1..Lmax, so 2L <= 128
        float ahi = abuf[prev][2 * L];
        float alo = abuf[prev][2 * L - 1];
        float m = fmaxf(ahi, alo);
        out[b] = -(m + logf(expf(ahi - m) + expf(alo - m)));
    }
}

// ---------------------------------------------------------------------------
extern "C" void kernel_launch(void* const* d_in, const int* in_sizes, int n_in,
                              void* d_out, int out_size, void* d_ws, size_t ws_size,
                              hipStream_t stream)
{
    (void)in_sizes; (void)n_in; (void)out_size; (void)ws_size;

    const int*   y_true = (const int*)  d_in[0];   // [B, Lmax]
    const float* y_pred = (const float*)d_in[1];   // [B, T, C]
    const int*   y_len  = (const int*)  d_in[2];   // [B, 1]

    float* lp  = (float*)d_ws;                     // [B*T*S] f32 = 33.8 MB scratch
    float* out = (float*)d_out;                    // [B, 1]

    const size_t shmem = 16 * 1024 * sizeof(float);            // 64 KB staging
    ctc_rowprep<<<dim3((B_ * T_) / 16), dim3(256), shmem, stream>>>(y_pred, y_true, lp);
    ctc_dp     <<<dim3(B_),             dim3(160), 0,     stream>>>(lp, y_true, y_len, out);
}
